// _MultiHeadAttention_13615046329068
// MI455X (gfx1250) — matrix operand order repaired and checked by reading, not compiled, not run on hardware
//
#include <hip/hip_runtime.h>
#include <hip/hip_bf16.h>

typedef __attribute__((ext_vector_type(16))) _Float16 v16h;
typedef __attribute__((ext_vector_type(8)))  _Float16 v8h;
typedef __attribute__((ext_vector_type(8)))  float    v8f;

#define B_   8
#define S_   1024
#define D_   1024
#define H_   16
#define DH_  64
#define BS_  (B_ * S_)
#define LDT  40   // LDS row stride in halfs (80B) to stagger banks

// ---------------- WMMA operand loaders (per CDNA5 ISA 7.12.2 layouts) -------

__device__ __forceinline__ v16h lds_load_a(const _Float16* base, int row, int lane) {
  const int hi = (lane >> 4) & 1;
  const _Float16* p = base + row * LDT;
  v8h x0 = *(const v8h*)(p + hi * 8);        // K = hi*8 .. hi*8+7
  v8h x1 = *(const v8h*)(p + 16 + hi * 8);   // K = 16+hi*8 .. 16+hi*8+7
  return __builtin_shufflevector(x0, x1, 0,1,2,3,4,5,6,7,8,9,10,11,12,13,14,15);
}

__device__ __forceinline__ v16h lds_load_b(const _Float16* base, int row, int lane) {
  const int hi = (lane >> 4) & 1;
  const _Float16* p = base + row * LDT;
  v8h x0 = *(const v8h*)(p + hi * 8);        // K = hi*8 .. hi*8+7
  v8h x1 = *(const v8h*)(p + 16 + hi * 8);   // K = 16+hi*8 .. 16+hi*8+7
  return __builtin_shufflevector(x0, x1, 0,1,2,3,4,5,6,7,8,9,10,11,12,13,14,15);
}

__device__ __forceinline__ void mma4(const _Float16* As, const _Float16* Bs,
                                     int wm, int wn, int lane, v8f acc[2][2]) {
  const int ln = lane & 15;
  v16h a0 = lds_load_a(As, wm + ln, lane);
  v16h a1 = lds_load_a(As, wm + 16 + ln, lane);
  v16h b0 = lds_load_b(Bs, wn + ln, lane);
  v16h b1 = lds_load_b(Bs, wn + 16 + ln, lane);
  acc[0][0] = __builtin_amdgcn_wmma_f32_16x16x32_f16(false, a0, false, b0, (short)0, acc[0][0], false, false);
  acc[0][1] = __builtin_amdgcn_wmma_f32_16x16x32_f16(false, a0, false, b1, (short)0, acc[0][1], false, false);
  acc[1][0] = __builtin_amdgcn_wmma_f32_16x16x32_f16(false, a1, false, b0, (short)0, acc[1][0], false, false);
  acc[1][1] = __builtin_amdgcn_wmma_f32_16x16x32_f16(false, a1, false, b1, (short)0, acc[1][1], false, false);
}

__device__ __forceinline__ void cvt_store16(_Float16* dst, const float* src) {
  float4 f0 = *(const float4*)(src + 0);
  float4 f1 = *(const float4*)(src + 4);
  float4 f2 = *(const float4*)(src + 8);
  float4 f3 = *(const float4*)(src + 12);
  v8h h0, h1;
  h0[0]=(_Float16)f0.x; h0[1]=(_Float16)f0.y; h0[2]=(_Float16)f0.z; h0[3]=(_Float16)f0.w;
  h0[4]=(_Float16)f1.x; h0[5]=(_Float16)f1.y; h0[6]=(_Float16)f1.z; h0[7]=(_Float16)f1.w;
  h1[0]=(_Float16)f2.x; h1[1]=(_Float16)f2.y; h1[2]=(_Float16)f2.z; h1[3]=(_Float16)f2.w;
  h1[4]=(_Float16)f3.x; h1[5]=(_Float16)f3.y; h1[6]=(_Float16)f3.z; h1[7]=(_Float16)f3.w;
  *(v8h*)(dst)     = h0;
  *(v8h*)(dst + 8) = h1;
}

// ---------------- Stage 1: weight transpose + f16 convert --------------------

__global__ __launch_bounds__(256)
void k_w_transpose(const float* __restrict__ W, _Float16* __restrict__ WT) {
  int idx = blockIdx.x * 256 + threadIdx.x;   // 1024*1024 elements
  int k = idx >> 10;
  int n = idx & 1023;
  WT[(size_t)n * D_ + k] = (_Float16)W[(size_t)k * D_ + n];
}

// ---------------- Stage 2: QKV projection (f32 A converted in staging) -------
// C(BS x 1024) = X @ W + bias, written head-split f16 (B,H,S,64)

__global__ __launch_bounds__(256)
void k_gemm_qkv(const float* __restrict__ X, const _Float16* __restrict__ WT,
                const float* __restrict__ bias, _Float16* __restrict__ Out) {
  __shared__ _Float16 As[128 * LDT];
  __shared__ _Float16 Bs[64 * LDT];
  const int t = threadIdx.x, lane = t & 31, wid = t >> 5;
  const int m0 = blockIdx.y * 128, n0 = blockIdx.x * 64;
  const int wm = (wid >> 1) * 32, wn = (wid & 1) * 32;
  const int ar = t >> 1, ac = (t & 1) * 16;   // A: 128 x 32
  const int br = t >> 2, bc = (t & 3) * 8;    // B: 64 x 32
  v8f acc[2][2] = {};
  for (int k0 = 0; k0 < D_; k0 += 32) {
    cvt_store16(&As[ar * LDT + ac], X + (size_t)(m0 + ar) * D_ + k0 + ac);
    *(v8h*)(&Bs[br * LDT + bc]) = *(const v8h*)(WT + (size_t)(n0 + br) * D_ + k0 + bc);
    __syncthreads();
    mma4(As, Bs, wm, wn, lane, acc);
    __syncthreads();
  }
  const int hi = lane >> 4, ln = lane & 15;
  for (int mt = 0; mt < 2; ++mt)
    for (int nt = 0; nt < 2; ++nt) {
      const int col = n0 + wn + nt * 16 + ln;
      const float bv = bias[col];
      const int h = col >> 6, dh = col & 63;
      for (int i = 0; i < 8; ++i) {
        const int row = m0 + wm + mt * 16 + hi * 8 + i;
        const int b = row >> 10, s = row & 1023;
        Out[((size_t)((b * H_ + h) * S_ + s)) * DH_ + dh] = (_Float16)(acc[mt][nt][i] + bv);
      }
    }
}

// ---------------- Stage 3: scores = Q @ K^T / sqrt(dk), masked, f32 ----------

__global__ __launch_bounds__(256)
void k_scores(const _Float16* __restrict__ Qh, const _Float16* __restrict__ Kh,
              const unsigned char* __restrict__ mask, float* __restrict__ attn) {
  __shared__ _Float16 As[128 * LDT];
  __shared__ _Float16 Bs[64 * LDT];
  const int t = threadIdx.x, lane = t & 31, wid = t >> 5;
  const int z = blockIdx.z;                 // z = h*B + b
  const int b = z & 7, h = z >> 3;
  const int m0 = blockIdx.y * 128, n0 = blockIdx.x * 64;
  const size_t base = (size_t)(b * H_ + h) * S_ * DH_;
  const _Float16* Q = Qh + base;
  const _Float16* K = Kh + base;
  const int wm = (wid >> 1) * 32, wn = (wid & 1) * 32;
  const int ar = t >> 1, ac = (t & 1) * 16;
  const int br = t >> 2, bc = (t & 3) * 8;
  v8f acc[2][2] = {};
  for (int k0 = 0; k0 < DH_; k0 += 32) {
    *(v8h*)(&As[ar * LDT + ac])     = *(const v8h*)(Q + (size_t)(m0 + ar) * DH_ + k0 + ac);
    *(v8h*)(&As[ar * LDT + ac + 8]) = *(const v8h*)(Q + (size_t)(m0 + ar) * DH_ + k0 + ac + 8);
    *(v8h*)(&Bs[br * LDT + bc])     = *(const v8h*)(K + (size_t)(n0 + br) * DH_ + k0 + bc);
    __syncthreads();
    mma4(As, Bs, wm, wn, lane, acc);
    __syncthreads();
  }
  const int hi = lane >> 4, ln = lane & 15;
  float* out = attn + (size_t)z * S_ * S_;
  const unsigned char* mrow = mask + (size_t)b * S_ * S_;
  for (int mt = 0; mt < 2; ++mt)
    for (int nt = 0; nt < 2; ++nt) {
      const int sk = n0 + wn + nt * 16 + ln;
      for (int i = 0; i < 8; ++i) {
        const int sq = m0 + wm + mt * 16 + hi * 8 + i;
        float vs = acc[mt][nt][i] * 0.125f;          // 1/sqrt(64)
        if (mrow[(size_t)sq * S_ + sk]) vs = -__builtin_inff();
        out[(size_t)sq * S_ + sk] = vs;
      }
    }
}

// ---------------- Stage 4: row softmax over Lk, in place ---------------------

__global__ __launch_bounds__(256)
void k_softmax(float* __restrict__ attn) {
  __shared__ float red[256];
  const int t = threadIdx.x;
  float* p = attn + (size_t)blockIdx.x * S_;
  float x[4]; float mx = -__builtin_inff();
  for (int j = 0; j < 4; ++j) { x[j] = p[t + 256 * j]; mx = fmaxf(mx, x[j]); }
  red[t] = mx; __syncthreads();
  for (int o = 128; o; o >>= 1) { if (t < o) red[t] = fmaxf(red[t], red[t + o]); __syncthreads(); }
  mx = red[0]; __syncthreads();
  float sum = 0.f;
  for (int j = 0; j < 4; ++j) { x[j] = __expf(x[j] - mx); sum += x[j]; }
  red[t] = sum; __syncthreads();
  for (int o = 128; o; o >>= 1) { if (t < o) red[t] += red[t + o]; __syncthreads(); }
  const float inv = 1.f / red[0];
  for (int j = 0; j < 4; ++j) p[t + 256 * j] = x[j] * inv;
}

// ---------------- Stage 5: ctx = attn @ V (f32 attn -> f16 in staging) -------

__global__ __launch_bounds__(256)
void k_attn_v(const float* __restrict__ attn, const _Float16* __restrict__ Vh,
              _Float16* __restrict__ Ctx) {
  __shared__ _Float16 As[128 * LDT];
  __shared__ _Float16 Bs[64 * LDT];
  const int t = threadIdx.x, lane = t & 31, wid = t >> 5;
  const int z = blockIdx.y;                 // z = h*B + b
  const int b = z & 7, h = z >> 3;
  const int m0 = blockIdx.x * 128;
  const float* A = attn + (size_t)z * S_ * S_;
  const _Float16* V = Vh + (size_t)(b * H_ + h) * S_ * DH_;
  const int wm = (wid >> 1) * 32, wn = (wid & 1) * 32;
  const int ar = t >> 1, ac = (t & 1) * 16;
  const int vr = t >> 3, vc = (t & 7) * 8;  // V tile: 32 k-rows x 64 dh-cols, transposed into Bs
  v8f acc[2][2] = {};
  for (int k0 = 0; k0 < S_; k0 += 32) {
    cvt_store16(&As[ar * LDT + ac], A + (size_t)(m0 + ar) * S_ + k0 + ac);
    v8h vv = *(const v8h*)(V + (size_t)(k0 + vr) * DH_ + vc);
#pragma unroll
    for (int j = 0; j < 8; ++j) Bs[(vc + j) * LDT + vr] = vv[j];
    __syncthreads();
    mma4(As, Bs, wm, wn, lane, acc);
    __syncthreads();
  }
  const int hi = lane >> 4, ln = lane & 15;
  for (int mt = 0; mt < 2; ++mt)
    for (int nt = 0; nt < 2; ++nt) {
      const int dh = wn + nt * 16 + ln;     // 0..63
      for (int i = 0; i < 8; ++i) {
        const int sq = m0 + wm + mt * 16 + hi * 8 + i;
        Ctx[(size_t)(b * S_ + sq) * D_ + h * DH_ + dh] = (_Float16)acc[mt][nt][i];
      }
    }
}

// ---------------- Stage 6: out-proj + bias + residual (pre-LN, f32) ----------

__global__ __launch_bounds__(256)
void k_gemm_out(const _Float16* __restrict__ Ctx, const _Float16* __restrict__ WoT,
                const float* __restrict__ bo, const float* __restrict__ resid,
                float* __restrict__ Y) {
  __shared__ _Float16 As[128 * LDT];
  __shared__ _Float16 Bs[64 * LDT];
  const int t = threadIdx.x, lane = t & 31, wid = t >> 5;
  const int m0 = blockIdx.y * 128, n0 = blockIdx.x * 64;
  const int wm = (wid >> 1) * 32, wn = (wid & 1) * 32;
  const int ar = t >> 1, ac = (t & 1) * 16;
  const int br = t >> 2, bc = (t & 3) * 8;
  v8f acc[2][2] = {};
  for (int k0 = 0; k0 < D_; k0 += 32) {
    *(v8h*)(&As[ar * LDT + ac])     = *(const v8h*)(Ctx + (size_t)(m0 + ar) * D_ + k0 + ac);
    *(v8h*)(&As[ar * LDT + ac + 8]) = *(const v8h*)(Ctx + (size_t)(m0 + ar) * D_ + k0 + ac + 8);
    *(v8h*)(&Bs[br * LDT + bc])     = *(const v8h*)(WoT + (size_t)(n0 + br) * D_ + k0 + bc);
    __syncthreads();
    mma4(As, Bs, wm, wn, lane, acc);
    __syncthreads();
  }
  const int hi = lane >> 4, ln = lane & 15;
  for (int mt = 0; mt < 2; ++mt)
    for (int nt = 0; nt < 2; ++nt) {
      const int col = n0 + wn + nt * 16 + ln;
      const float bv = bo[col];
      for (int i = 0; i < 8; ++i) {
        const int row = m0 + wm + mt * 16 + hi * 8 + i;
        Y[(size_t)row * D_ + col] = acc[mt][nt][i] + bv + resid[(size_t)row * D_ + col];
      }
    }
}

// ---------------- Stage 7: LayerNorm rows of 1024, in place ------------------

__global__ __launch_bounds__(256)
void k_layernorm(float* __restrict__ Y, const float* __restrict__ gamma,
                 const float* __restrict__ beta) {
  __shared__ float red[256];
  const int t = threadIdx.x;
  float* p = Y + (size_t)blockIdx.x * D_;
  float x[4]; float s = 0.f, s2 = 0.f;
  for (int j = 0; j < 4; ++j) { x[j] = p[t + 256 * j]; s += x[j]; s2 += x[j] * x[j]; }
  red[t] = s; __syncthreads();
  for (int o = 128; o; o >>= 1) { if (t < o) red[t] += red[t + o]; __syncthreads(); }
  const float mean = red[0] * (1.f / 1024.f);
  __syncthreads();
  red[t] = s2; __syncthreads();
  for (int o = 128; o; o >>= 1) { if (t < o) red[t] += red[t + o]; __syncthreads(); }
  const float var = red[0] * (1.f / 1024.f) - mean * mean;
  const float inv = rsqrtf(var + 1e-5f);
  for (int j = 0; j < 4; ++j) {
    const int c = t + 256 * j;
    p[c] = (x[j] - mean) * inv * gamma[c] + beta[c];
  }
}

// ---------------- launch -----------------------------------------------------

extern "C" void kernel_launch(void* const* d_in, const int* in_sizes, int n_in,
                              void* d_out, int out_size, void* d_ws, size_t ws_size,
                              hipStream_t stream) {
  const float* q   = (const float*)d_in[0];
  const float* k   = (const float*)d_in[1];
  const float* v   = (const float*)d_in[2];
  const unsigned char* mask = (const unsigned char*)d_in[3];
  const float* Wq  = (const float*)d_in[4];
  const float* bq  = (const float*)d_in[5];
  const float* Wk  = (const float*)d_in[6];
  const float* bk  = (const float*)d_in[7];
  const float* Wv  = (const float*)d_in[8];
  const float* bv  = (const float*)d_in[9];
  const float* Wo  = (const float*)d_in[10];
  const float* bo  = (const float*)d_in[11];
  const float* gamma = (const float*)d_in[12];
  const float* beta  = (const float*)d_in[13];

  float* y    = (float*)d_out;
  float* attn = y + (size_t)BS_ * D_;

  _Float16* WqT = (_Float16*)d_ws;
  _Float16* WkT = WqT + (size_t)D_ * D_;
  _Float16* WvT = WkT + (size_t)D_ * D_;
  _Float16* WoT = WvT + (size_t)D_ * D_;
  _Float16* Qh  = WoT + (size_t)D_ * D_;
  _Float16* Kh  = Qh  + (size_t)BS_ * D_;
  _Float16* Vh  = Kh  + (size_t)BS_ * D_;
  _Float16* Ctx = Vh  + (size_t)BS_ * D_;

  dim3 blk(256);
  dim3 gW(D_ * D_ / 256);
  k_w_transpose<<<gW, blk, 0, stream>>>(Wq, WqT);
  k_w_transpose<<<gW, blk, 0, stream>>>(Wk, WkT);
  k_w_transpose<<<gW, blk, 0, stream>>>(Wv, WvT);
  k_w_transpose<<<gW, blk, 0, stream>>>(Wo, WoT);

  dim3 gP(D_ / 64, BS_ / 128);
  k_gemm_qkv<<<gP, blk, 0, stream>>>(q, WqT, bq, Qh);
  k_gemm_qkv<<<gP, blk, 0, stream>>>(k, WkT, bk, Kh);
  k_gemm_qkv<<<gP, blk, 0, stream>>>(v, WvT, bv, Vh);

  k_scores<<<dim3(S_ / 64, S_ / 128, H_ * B_), blk, 0, stream>>>(Qh, Kh, mask, attn);
  k_softmax<<<dim3(H_ * B_ * S_), blk, 0, stream>>>(attn);
  k_attn_v<<<dim3(S_ / 128, H_ * B_), blk, 0, stream>>>(attn, Vh, Ctx);
  k_gemm_out<<<gP, blk, 0, stream>>>(Ctx, WoT, bo, q, y);
  k_layernorm<<<dim3(BS_), blk, 0, stream>>>(y, gamma, beta);
}
